// GraphNetwork_43018392436800
// MI455X (gfx1250) — matrix operand order repaired and checked by reading, not compiled, not run on hardware
//
#include <hip/hip_runtime.h>
#include <hip/hip_bf16.h>

#define Bsz   32
#define Kc    8
#define Sl    512
#define Dd    600
#define Hh    600
#define NNODE 320      // 2*B + B*K
#define KP1   5408     // 9*600 = 5400 padded to mult of 32
#define NP    608      // 600 padded to mult of 16 (also mult of 32 for K reuse)
#define KP2   608      // K for GAT GEMM (out1 padded cols are zero)

typedef __attribute__((ext_vector_type(16))) _Float16 v16h;
typedef __attribute__((ext_vector_type(8)))  _Float16 v8h;
typedef __attribute__((ext_vector_type(8)))  float    v8f;
typedef __attribute__((ext_vector_type(4)))  float    v4f;

__device__ __forceinline__ const float* node_row(const float* t, const float* e,
                                                 const float* c, int n) {
  if (n < Bsz)      return t + (size_t)n * Dd;
  if (n < 2 * Bsz)  return e + (size_t)(n - Bsz) * Dd;
  return c + (size_t)(n - 2 * Bsz) * Dd;
}

// ---------------- K1: Wcat^T (hi/lo f16), layout [o][k], k = r*600+i, r=8 -> root
__global__ void k_build_w(const float* __restrict__ basis, const float* __restrict__ comp,
                          const float* __restrict__ root,
                          _Float16* __restrict__ wt_h, _Float16* __restrict__ wt_l) {
  int t = blockIdx.x * blockDim.x + threadIdx.x;
  if (t >= Dd * Hh) return;
  int i = t / Hh, o = t % Hh;        // o fastest -> coalesced basis reads
  float acc[8];
#pragma unroll
  for (int r = 0; r < 8; ++r) acc[r] = 0.f;
  for (int b = 0; b < 30; ++b) {
    float v = basis[((size_t)b * Dd + i) * Hh + o];
#pragma unroll
    for (int r = 0; r < 8; ++r) acc[r] += comp[r * 30 + b] * v;
  }
#pragma unroll
  for (int r = 0; r < 8; ++r) {
    int k = r * Dd + i;
    _Float16 hi = (_Float16)acc[r];
    wt_h[(size_t)o * KP1 + k] = hi;
    wt_l[(size_t)o * KP1 + k] = (_Float16)(acc[r] - (float)hi);
  }
  float rv = root[(size_t)i * Hh + o];
  _Float16 rhi = (_Float16)rv;
  wt_h[(size_t)o * KP1 + 8 * Dd + i] = rhi;
  wt_l[(size_t)o * KP1 + 8 * Dd + i] = (_Float16)(rv - (float)rhi);
}

// ---------------- K2: per-(dst,type) neighbor sums + counts (f32 atomics)
__global__ void k_edge_agg(const int* __restrict__ src, const int* __restrict__ dst,
                           const int* __restrict__ etype, int E,
                           const float* __restrict__ t, const float* __restrict__ e,
                           const float* __restrict__ c,
                           float* __restrict__ xbar, float* __restrict__ cnt) {
  int ed = blockIdx.x;
  if (ed >= E) return;
  int s = src[ed], d = dst[ed], r = etype[ed];
  const float* row = node_row(t, e, c, s);
  float* outp = xbar + ((size_t)d * 8 + r) * Dd;
  for (int j = threadIdx.x; j < Dd; j += blockDim.x) atomicAdd(outp + j, row[j]);
  if (threadIdx.x == 0) atomicAdd(cnt + d * 8 + r, 1.0f);
}

// ---------------- K3: Xcat (hi/lo), [n][KP1]: k<4800 mean msgs, 4800..5399 = x, rest 0
__global__ void k_build_xcat(const float* __restrict__ xbar, const float* __restrict__ cnt,
                             const float* __restrict__ t, const float* __restrict__ e,
                             const float* __restrict__ c,
                             _Float16* __restrict__ xh, _Float16* __restrict__ xl) {
  int idx = blockIdx.x * blockDim.x + threadIdx.x;
  if (idx >= NNODE * KP1) return;
  int n = idx / KP1, k = idx % KP1;
  float v = 0.f;
  if (k < 8 * Dd) {
    int r = k / Dd, i = k % Dd;
    v = xbar[((size_t)n * 8 + r) * Dd + i] / fmaxf(cnt[n * 8 + r], 1.0f);
  } else if (k < 9 * Dd) {
    v = node_row(t, e, c, n)[k - 8 * Dd];
  }
  _Float16 hi = (_Float16)v;
  xh[idx] = hi;
  xl[idx] = (_Float16)(v - (float)hi);
}

// ---------------- K4: hi/lo-split f16 WMMA GEMM: C[M][Np] = A[M][Kp] * B^T[Np][Kp] (+bias)
__global__ void k_gemm_hilo(const _Float16* __restrict__ Ah, const _Float16* __restrict__ Al,
                            const _Float16* __restrict__ Bh, const _Float16* __restrict__ Bl,
                            const float* __restrict__ bias, int biasN,
                            float* __restrict__ C, int M, int Kp, int Np) {
  int wave = blockIdx.x * (blockDim.x >> 5) + (threadIdx.x >> 5);
  int tw = Np >> 4;
  int ntiles = (M >> 4) * tw;
  if (wave >= ntiles) return;               // uniform per-wave: EXEC stays all-ones
  int mt = wave / tw, ot = wave % tw;
  int lane = threadIdx.x & 31;
  int mr = lane & 15, h = lane >> 4;
  const _Float16* arh = Ah + (size_t)(mt * 16 + mr) * Kp;
  const _Float16* arl = Al + (size_t)(mt * 16 + mr) * Kp;
  const _Float16* brh = Bh + (size_t)(ot * 16 + mr) * Kp;   // row index = output col
  const _Float16* brl = Bl + (size_t)(ot * 16 + mr) * Kp;
  v8f acc = {0.f, 0.f, 0.f, 0.f, 0.f, 0.f, 0.f, 0.f};
  for (int kb = 0; kb < Kp; kb += 32) {
    if (kb + 64 < Kp) {
      __builtin_prefetch(arh + kb + 64, 0, 1);
      __builtin_prefetch(brh + kb + 64, 0, 1);
    }
    // A layout: j=0..7 -> K=kb+8h+j ; j=8..15 -> K=kb+16+8h+(j-8)
    v8h a0h = *(const v8h*)(arh + kb + h * 8);
    v8h a1h = *(const v8h*)(arh + kb + 16 + h * 8);
    v8h a0l = *(const v8h*)(arl + kb + h * 8);
    v8h a1l = *(const v8h*)(arl + kb + 16 + h * 8);
    // B layout: j=0..7 -> K=kb+8h+j ; j=8..15 -> K=kb+16+8h+(j-8)
    v8h b0h = *(const v8h*)(brh + kb + h * 8);
    v8h b1h = *(const v8h*)(brh + kb + 16 + h * 8);
    v8h b0l = *(const v8h*)(brl + kb + h * 8);
    v8h b1l = *(const v8h*)(brl + kb + 16 + h * 8);
    v16h a_hi = __builtin_shufflevector(a0h, a1h, 0,1,2,3,4,5,6,7,8,9,10,11,12,13,14,15);
    v16h a_lo = __builtin_shufflevector(a0l, a1l, 0,1,2,3,4,5,6,7,8,9,10,11,12,13,14,15);
    v16h b_hi = __builtin_shufflevector(b0h, b1h, 0,1,2,3,4,5,6,7,8,9,10,11,12,13,14,15);
    v16h b_lo = __builtin_shufflevector(b0l, b1l, 0,1,2,3,4,5,6,7,8,9,10,11,12,13,14,15);
    acc = __builtin_amdgcn_wmma_f32_16x16x32_f16(false, a_hi, false, b_hi, (short)0, acc, false, false);
    acc = __builtin_amdgcn_wmma_f32_16x16x32_f16(false, a_hi, false, b_lo, (short)0, acc, false, false);
    acc = __builtin_amdgcn_wmma_f32_16x16x32_f16(false, a_lo, false, b_hi, (short)0, acc, false, false);
  }
  int o = ot * 16 + mr;
  float bv = (bias != nullptr && o < biasN) ? bias[o] : 0.0f;
#pragma unroll
  for (int v = 0; v < 8; ++v) {         // D layout: VGPR v -> row v + 8h, col = lane&15
    int row = mt * 16 + v + 8 * h;
    C[(size_t)row * Np + o] = acc[v] + bv;
  }
}

// ---------------- K5: generic f32 -> (hi,lo) f16 split
__global__ void k_split_f16(const float* __restrict__ in, _Float16* __restrict__ oh,
                            _Float16* __restrict__ ol, int total) {
  int t = blockIdx.x * blockDim.x + threadIdx.x;
  if (t >= total) return;
  float v = in[t];
  _Float16 hi = (_Float16)v;
  oh[t] = hi;
  ol[t] = (_Float16)(v - (float)hi);
}

// ---------------- K6: w_gat^T (hi/lo), [o][KP2]
__global__ void k_build_wg(const float* __restrict__ wgat,
                           _Float16* __restrict__ wh, _Float16* __restrict__ wl) {
  int t = blockIdx.x * blockDim.x + threadIdx.x;
  if (t >= Dd * Hh) return;
  int i = t / Hh, o = t % Hh;
  float v = wgat[(size_t)i * Hh + o];
  _Float16 hi = (_Float16)v;
  wh[(size_t)o * KP2 + i] = hi;
  wl[(size_t)o * KP2 + i] = (_Float16)(v - (float)hi);
}

// ---------------- K7: attention scalar dots  s_src[n]=xl[n].att_src, s_dst likewise
__global__ void k_dots(const float* __restrict__ xl, const float* __restrict__ as,
                       const float* __restrict__ ad, float* __restrict__ ssrc,
                       float* __restrict__ sdst) {
  int n = blockIdx.x;
  __shared__ float r1[256], r2[256];
  float p1 = 0.f, p2 = 0.f;
  for (int h = threadIdx.x; h < Hh; h += blockDim.x) {
    float v = xl[(size_t)n * NP + h];
    p1 += v * as[h];
    p2 += v * ad[h];
  }
  r1[threadIdx.x] = p1; r2[threadIdx.x] = p2;
  __syncthreads();
  for (int s = 128; s > 0; s >>= 1) {
    if (threadIdx.x < s) { r1[threadIdx.x] += r1[threadIdx.x + s]; r2[threadIdx.x] += r2[threadIdx.x + s]; }
    __syncthreads();
  }
  if (threadIdx.x == 0) { ssrc[n] = r1[0]; sdst[n] = r2[0]; }
}

// ---------------- K8: per-cause-node GAT softmax + aggregation (cause nodes only)
__global__ void k_gat_agg(const int* __restrict__ src, const int* __restrict__ dst, int E,
                          const float* __restrict__ xl, const float* __restrict__ ssrc,
                          const float* __restrict__ sdst, const float* __restrict__ bias2,
                          float* __restrict__ out2c) {
  int n = 2 * Bsz + blockIdx.x;             // cause nodes 64..319
  __shared__ int   scnt;
  __shared__ int   lsrc[64];
  __shared__ float lval[64];
  __shared__ float lcoef[64];
  __shared__ float selfcoef;
  int tid = threadIdx.x;
  if (tid == 0) scnt = 0;
  __syncthreads();
  for (int eidx = tid; eidx < E; eidx += blockDim.x) {
    if (dst[eidx] == n) {
      int slot = atomicAdd(&scnt, 1);
      if (slot < 63) lsrc[slot] = src[eidx];
    }
  }
  __syncthreads();
  int cnt = min(scnt, 63);
  float sdn = sdst[n];
  if (tid < cnt) {
    float v = ssrc[lsrc[tid]] + sdn;
    lval[tid] = v >= 0.f ? v : 0.2f * v;
  }
  if (tid == 0) {
    float v = ssrc[n] + sdn;                // self loop
    lval[63] = v >= 0.f ? v : 0.2f * v;
  }
  __syncthreads();
  if (tid == 0) {
    float m = lval[63];
    for (int l = 0; l < cnt; ++l) m = fmaxf(m, lval[l]);
    float z = __expf(lval[63] - m);
    for (int l = 0; l < cnt; ++l) z += __expf(lval[l] - m);
    for (int l = 0; l < cnt; ++l) lcoef[l] = __expf(lval[l] - m) / z;
    selfcoef = __expf(lval[63] - m) / z;
  }
  __syncthreads();
  float a0 = 0.f, a1 = 0.f, a2 = 0.f;
  for (int l = 0; l < cnt; ++l) {
    const float* row = xl + (size_t)lsrc[l] * NP;
    float cf = lcoef[l];
    a0 += cf * row[tid];
    if (tid + 256 < Hh) a1 += cf * row[tid + 256];
    if (tid + 512 < Hh) a2 += cf * row[tid + 512];
  }
  {
    const float* row = xl + (size_t)n * NP;
    a0 += selfcoef * row[tid];
    if (tid + 256 < Hh) a1 += selfcoef * row[tid + 256];
    if (tid + 512 < Hh) a2 += selfcoef * row[tid + 512];
  }
  float* orow = out2c + (size_t)blockIdx.x * Hh;
  orow[tid] = a0 + bias2[tid];
  if (tid + 256 < Hh) orow[tid + 256] = a1 + bias2[tid + 256];
  if (tid + 512 < Hh) orow[tid + 512] = a2 + bias2[tid + 512];
}

// ---------------- K9: broadcast 256x600 -> 256x512x600 with non-temporal float4 stores
__global__ void k_bcast(const float* __restrict__ out2c, float* __restrict__ out) {
  int cblk = blockIdx.x;
  int sg = blockIdx.y;                      // 8 groups of 64 rows
  __shared__ v4f row[Hh / 4];
  int tid = threadIdx.x;
  if (tid < Hh / 4) row[tid] = ((const v4f*)(out2c + (size_t)cblk * Hh))[tid];
  __syncthreads();
  float* base = out + ((size_t)cblk * Sl + (size_t)sg * 64) * Hh;
  const int QP = Hh / 4;                    // 150 float4 per row
  for (int idx = tid; idx < 64 * QP; idx += blockDim.x) {
    int s = idx / QP, q = idx - s * QP;
    __builtin_nontemporal_store(row[q], (v4f*)(base + (size_t)s * Hh) + q);
  }
}

extern "C" void kernel_launch(void* const* d_in, const int* in_sizes, int n_in,
                              void* d_out, int out_size, void* d_ws, size_t ws_size,
                              hipStream_t stream) {
  const float* target = (const float*)d_in[0];
  const float* cause  = (const float*)d_in[1];
  const float* emo    = (const float*)d_in[2];
  /* d_in[3] word_node: shape-only, never read (saves 315 MB of traffic) */
  const int*   eidx   = (const int*)d_in[4];
  const int*   etype  = (const int*)d_in[5];
  const float* basis  = (const float*)d_in[6];
  const float* comp   = (const float*)d_in[7];
  const float* root   = (const float*)d_in[8];
  const float* bias1  = (const float*)d_in[9];
  const float* wgat   = (const float*)d_in[10];
  const float* attsrc = (const float*)d_in[11];
  const float* attdst = (const float*)d_in[12];
  const float* bias2  = (const float*)d_in[13];
  int E = in_sizes[5];
  const int* srcv = eidx;
  const int* dstv = eidx + E;

  // workspace bump allocator (256B aligned)
  char* ws = (char*)d_ws;
  size_t off = 0;
  auto alloc = [&](size_t bytes) -> char* {
    char* p = ws + off;
    off += (bytes + 255) & ~(size_t)255;
    return p;
  };
  _Float16* wt_h  = (_Float16*)alloc((size_t)NP * KP1 * 2);
  _Float16* wt_l  = (_Float16*)alloc((size_t)NP * KP1 * 2);
  _Float16* xc_h  = (_Float16*)alloc((size_t)NNODE * KP1 * 2);
  _Float16* xc_l  = (_Float16*)alloc((size_t)NNODE * KP1 * 2);
  float*    xbar  = (float*)   alloc((size_t)NNODE * 8 * Dd * 4);
  float*    cntb  = (float*)   alloc((size_t)NNODE * 8 * 4);
  float*    out1  = (float*)   alloc((size_t)NNODE * NP * 4);
  _Float16* xg_h  = (_Float16*)alloc((size_t)NNODE * KP2 * 2);
  _Float16* xg_l  = (_Float16*)alloc((size_t)NNODE * KP2 * 2);
  _Float16* wg_h  = (_Float16*)alloc((size_t)NP * KP2 * 2);
  _Float16* wg_l  = (_Float16*)alloc((size_t)NP * KP2 * 2);
  float*    xlbuf = (float*)   alloc((size_t)NNODE * NP * 4);
  float*    ssrc  = (float*)   alloc((size_t)NNODE * 4);
  float*    sdst  = (float*)   alloc((size_t)NNODE * 4);
  float*    out2c = (float*)   alloc((size_t)(Bsz * Kc) * Hh * 4);

  // zero pads + atomic accumulators (must be re-done every call)
  hipMemsetAsync(d_ws, 0, off, stream);

  // 1) Wcat^T hi/lo
  k_build_w<<<(Dd * Hh + 255) / 256, 256, 0, stream>>>(basis, comp, root, wt_h, wt_l);
  // 2) edge mean aggregation
  k_edge_agg<<<E, 256, 0, stream>>>(srcv, dstv, etype, E, target, emo, cause, xbar, cntb);
  // 3) Xcat hi/lo
  k_build_xcat<<<(NNODE * KP1 + 255) / 256, 256, 0, stream>>>(xbar, cntb, target, emo, cause, xc_h, xc_l);
  // 4) out1 = Xcat @ Wcat + bias1   (M=320, K=5408, N=608) — WMMA
  {
    int ntiles = (NNODE / 16) * (NP / 16);          // 760
    int blocks = (ntiles + 7) / 8;                  // 8 waves/block
    k_gemm_hilo<<<blocks, 256, 0, stream>>>(xc_h, xc_l, wt_h, wt_l, bias1, Hh,
                                            out1, NNODE, KP1, NP);
  }
  // 5) split out1 -> f16 hi/lo (pad cols already zero)
  k_split_f16<<<(NNODE * NP + 255) / 256, 256, 0, stream>>>(out1, xg_h, xg_l, NNODE * NP);
  // 6) w_gat^T hi/lo
  k_build_wg<<<(Dd * Hh + 255) / 256, 256, 0, stream>>>(wgat, wg_h, wg_l);
  // 7) xl = out1 @ w_gat   (M=320, K=608, N=608) — WMMA
  {
    int ntiles = (NNODE / 16) * (NP / 16);
    int blocks = (ntiles + 7) / 8;
    k_gemm_hilo<<<blocks, 256, 0, stream>>>(xg_h, xg_l, wg_h, wg_l, nullptr, 0,
                                            xlbuf, NNODE, KP2, NP);
  }
  // 8) attention dot products
  k_dots<<<NNODE, 256, 0, stream>>>(xlbuf, attsrc, attdst, ssrc, sdst);
  // 9) GAT softmax + aggregate (cause nodes only)
  k_gat_agg<<<Bsz * Kc, 256, 0, stream>>>(srcv, dstv, E, xlbuf, ssrc, sdst, bias2, out2c);
  // 10) broadcast to (256, 512, 600) with streaming stores
  {
    dim3 grid(Bsz * Kc, Sl / 64);
    k_bcast<<<grid, 256, 0, stream>>>(out2c, (float*)d_out);
  }
}
